// SwinTransformerBlock_42322607735267
// MI455X (gfx1250) — matrix operand order repaired and checked by reading, not compiled, not run on hardware
//
#include <hip/hip_runtime.h>

typedef _Float16 v16h __attribute__((ext_vector_type(16)));
typedef _Float16 v8h  __attribute__((ext_vector_type(8)));
typedef float    v8f  __attribute__((ext_vector_type(8)));

// ---- static problem dims (Swin block) ----
#define BB      32
#define HH      56
#define WWID    56
#define CC      384
#define WIN     7
#define SHIFT_  3
#define HEADS   12
#define MLPD    1536
#define NTOK    49          // tokens per window
#define NWIN    64          // windows per image (8x8)
#define HD      32          // head dim
#define MTOT    100352      // BB*HH*WWID = 2048 windows * 49
#define SCALE_Q 0.17677669529663687f   // HD^-0.5

__device__ __forceinline__ v16h cat8(v8h lo, v8h hi) {
    return __builtin_shufflevector(lo, hi, 0,1,2,3,4,5,6,7,8,9,10,11,12,13,14,15);
}
__device__ __forceinline__ v8f wmma16(v16h a, v16h b, v8f c) {
    // v_wmma_f32_16x16x32_f16 (neg_a, A, neg_b, B, c_mod, C, reuse_a, reuse_b)
    return __builtin_amdgcn_wmma_f32_16x16x32_f16(false, a, false, b, (short)0, c, false, false);
}

// CDNA5 async global->LDS copy (16 bytes, per-lane addresses, ASYNCcnt-tracked)
__device__ __forceinline__ void async_ld_b128(const _Float16* g, _Float16* l) {
    unsigned ldsOff = (unsigned)(unsigned long long)
        (__attribute__((address_space(3))) _Float16*)l;
    unsigned long long ga = (unsigned long long)g;
    asm volatile("global_load_async_to_lds_b128 %0, %1, off"
                 :: "v"(ldsOff), "v"(ga) : "memory");
}
// wait until all of this wave's async transfers have completed
__device__ __forceinline__ void wait_async0() {
    asm volatile("s_wait_asynccnt 0x0" ::: "memory");
}
// partial wait: allow up to 4 (one prefetched tile) still in flight
__device__ __forceinline__ void wait_async4() {
    asm volatile("s_wait_asynccnt 0x4" ::: "memory");
}

// window-ordered row -> original token row (undo shift + window partition)
__device__ __forceinline__ int win_row_to_orig(int row) {
    int win = row / NTOK, n = row % NTOK;
    int b  = win >> 6;
    int wy = (win >> 3) & 7;
    int wx = win & 7;
    int hs = wy * WIN + n / WIN;
    int ws = wx * WIN + n % WIN;
    int h0 = hs + SHIFT_; if (h0 >= HH)   h0 -= HH;
    int w0 = ws + SHIFT_; if (w0 >= WWID) w0 -= WWID;
    return b * (HH * WWID) + h0 * WWID + w0;
}

// shifted-window region label for mask (static ATTN_MASK reconstruction)
__device__ __forceinline__ int swin_label(int wy, int wx, int t) {
    int sy = wy * WIN + t / WIN;
    int sx = wx * WIN + t % WIN;
    int ry = (sy < HH - WIN) ? 0 : ((sy < HH - SHIFT_) ? 1 : 2);
    int rx = (sx < WWID - WIN) ? 0 : ((sx < WWID - SHIFT_) ? 1 : 2);
    return ry * 3 + rx;
}

// ------------------------------------------------------------------
// Generic f16 WMMA GEMM: C[M,N] = A[M,K] @ B[K,N] (BT stored [N,K])
// WG tile 128x128, 8 waves (2x4), each wave 64x32 (4x2 accum tiles).
// K tiles double-buffered through the CDNA5 async global->LDS engine:
// prefetch tile k+1, s_wait_asynccnt<=4 to consume tile k (in-order
// completion per wave), so global traffic overlaps the WMMA work.
// ------------------------------------------------------------------
enum { EPI_QKV = 0, EPI_PROJ = 1, EPI_MLP1 = 2, EPI_MLP2 = 3 };

template<int EPI>
__global__ __launch_bounds__(256) void gemm_wmma_kernel(
    const _Float16* __restrict__ A, const _Float16* __restrict__ BT,
    const float* __restrict__ bias, const float* __restrict__ resid,
    void* __restrict__ outp, int M, int Nn, int K)
{
    (void)M;
    __shared__ _Float16 As[2][128 * 32];
    __shared__ _Float16 Bs[2][128 * 32];

    const int tid   = threadIdx.x;
    const int lane  = tid & 31;
    const int wave  = tid >> 5;
    const int waveM = wave >> 2;    // 0..1 (64 rows each)
    const int waveN = wave & 3;     // 0..3 (32 cols each)
    const int m16   = lane & 15;
    const int kh    = lane >> 4;
    const int mBlock = blockIdx.y * 128;
    const int nBlock = blockIdx.x * 128;

    const int ldRow = tid >> 1;          // 0..127
    const int ldSeg = (tid & 1) * 16;    // 0 or 16 halfs

    // issue the 4 async b128 loads for K-tile starting at k0 into buffer b
    auto issue_tile = [&](int k0, int b) {
        const _Float16* ga = A + (size_t)(mBlock + ldRow) * K + k0 + ldSeg;
        async_ld_b128(ga,     &As[b][ldRow * 32 + ldSeg]);
        async_ld_b128(ga + 8, &As[b][ldRow * 32 + ldSeg + 8]);
        const _Float16* gb = BT + (size_t)(nBlock + ldRow) * K + k0 + ldSeg;
        async_ld_b128(gb,     &Bs[b][ldRow * 32 + ldSeg]);
        async_ld_b128(gb + 8, &Bs[b][ldRow * 32 + ldSeg + 8]);
    };

    const v8f z = {0.f,0.f,0.f,0.f,0.f,0.f,0.f,0.f};
    v8f acc[4][2];
    #pragma unroll
    for (int mt = 0; mt < 4; ++mt)
        #pragma unroll
        for (int nt = 0; nt < 2; ++nt) acc[mt][nt] = z;

    const int nK = K >> 5;          // K/32 tiles
    issue_tile(0, 0);               // prologue prefetch

    for (int kt = 0; kt < nK; ++kt) {
        const int cur = kt & 1;
        // prefetch next tile into the buffer everyone stopped reading at the
        // end-barrier of iteration kt-1; then wait for tile kt only.
        if (kt + 1 < nK) { issue_tile((kt + 1) << 5, cur ^ 1); wait_async4(); }
        else             { wait_async0(); }
        __syncthreads();            // all waves' tile-kt LDS writes visible

        v16h bfrag[2];
        #pragma unroll
        for (int nt = 0; nt < 2; ++nt) {
            int brow = waveN * 32 + nt * 16 + m16;
            v8h lo = *(const v8h*)&Bs[cur][brow * 32 + kh * 8];
            v8h hi = *(const v8h*)&Bs[cur][brow * 32 + 16 + kh * 8];
            bfrag[nt] = cat8(lo, hi);
        }
        #pragma unroll
        for (int mt = 0; mt < 4; ++mt) {
            int arow = waveM * 64 + mt * 16 + m16;
            v8h lo = *(const v8h*)&As[cur][arow * 32 + kh * 8];
            v8h hi = *(const v8h*)&As[cur][arow * 32 + 16 + kh * 8];
            v16h af = cat8(lo, hi);
            #pragma unroll
            for (int nt = 0; nt < 2; ++nt)
                acc[mt][nt] = wmma16(af, bfrag[nt], acc[mt][nt]);
        }
        __syncthreads();            // everyone done reading tile kt
    }

    // epilogue
    #pragma unroll
    for (int mt = 0; mt < 4; ++mt) {
        #pragma unroll
        for (int nt = 0; nt < 2; ++nt) {
            v8f c = acc[mt][nt];
            int gcol = nBlock + waveN * 32 + nt * 16 + m16;
            float bb = bias[gcol];
            #pragma unroll
            for (int r = 0; r < 8; ++r) {
                int grow = mBlock + waveM * 64 + mt * 16 + r + 8 * kh;
                float v = c[r] + bb;
                if constexpr (EPI == EPI_QKV) {
                    if (gcol < CC) v *= SCALE_Q;              // scale q (after bias)
                    ((_Float16*)outp)[(size_t)grow * Nn + gcol] = (_Float16)v;
                } else if constexpr (EPI == EPI_MLP1) {
                    v = 0.5f * v * (1.0f + erff(v * 0.70710678118654752f)); // exact gelu
                    ((_Float16*)outp)[(size_t)grow * Nn + gcol] = (_Float16)v;
                } else if constexpr (EPI == EPI_PROJ) {
                    size_t orow = (size_t)win_row_to_orig(grow); // window-reverse + unshift
                    ((float*)outp)[orow * CC + gcol] = resid[orow * CC + gcol] + v;
                } else { // EPI_MLP2
                    ((float*)outp)[(size_t)grow * CC + gcol] =
                        resid[(size_t)grow * CC + gcol] + v;
                }
            }
        }
    }
}

// ------------------------------------------------------------------
// Windowed attention: one wave per (window, head). N=49 padded to 64.
// S = Q@K^T (+rpb bias + shift mask), softmax, O = P@V.
// ------------------------------------------------------------------
__global__ __launch_bounds__(64) void attn_kernel(
    const _Float16* __restrict__ qkv,   // [MTOT, 1152] f16 (q pre-scaled)
    const float*    __restrict__ rpb,   // [169, 12]
    _Float16*       __restrict__ out)   // [MTOT, 384] f16
{
    __shared__ float    Sw_s[2][64 * 65];
    __shared__ _Float16 Pw_s[2][64 * 72];
    __shared__ _Float16 Vt_s[2][32 * 72];

    const int tid  = threadIdx.x;
    const int lane = tid & 31;
    const int wave = tid >> 5;
    const int head = blockIdx.y * 2 + wave;
    const int win  = blockIdx.x;
    const int wy   = (win >> 3) & 7;
    const int wx   = win & 7;
    const size_t qbase = (size_t)win * NTOK;
    const int m16 = lane & 15;
    const int kh  = lane >> 4;

    float*    Sw = Sw_s[wave];
    _Float16* Pw = Pw_s[wave];
    _Float16* Vt = Vt_s[wave];

    // stage V transposed into LDS, zero-pad K >= 49 (avoids garbage*0 NaN)
    for (int idx = lane; idx < 32 * 64; idx += 32) {
        int d = idx >> 6, kk = idx & 63;
        _Float16 val = (_Float16)0.f;
        if (kk < NTOK) val = qkv[(qbase + kk) * 1152 + 768 + head * HD + d];
        Vt[d * 72 + kk] = val;
    }

    const v8f z = {0.f,0.f,0.f,0.f,0.f,0.f,0.f,0.f};

    // K fragments (B operand): lane reads K-halfs kh*8..+7 and 16+kh*8..+7 of key row n
    v16h kf[4];
    #pragma unroll
    for (int nt = 0; nt < 4; ++nt) {
        int n = nt * 16 + m16; if (n > NTOK - 1) n = NTOK - 1;  // clamp pad rows
        const _Float16* p = qkv + (qbase + n) * 1152 + CC + head * HD + kh * 8;
        kf[nt] = cat8(*(const v8h*)p, *(const v8h*)(p + 16));
    }

    // S = Q @ K^T, one WMMA per 16x16 tile (K=HD=32 in one step)
    #pragma unroll
    for (int mt = 0; mt < 4; ++mt) {
        int i = mt * 16 + m16; int gi = (i > NTOK - 1) ? NTOK - 1 : i;
        const _Float16* p = qkv + (qbase + gi) * 1152 + head * HD + kh * 8;
        v16h qf = cat8(*(const v8h*)p, *(const v8h*)(p + 16));
        #pragma unroll
        for (int nt = 0; nt < 4; ++nt) {
            v8f c = wmma16(qf, kf[nt], z);
            int j = nt * 16 + m16;
            #pragma unroll
            for (int r = 0; r < 8; ++r) {
                int ii = mt * 16 + r + 8 * kh;
                float v = c[r];
                if (j >= NTOK) {
                    v = -1e30f;                 // mask padded keys hard
                } else if (ii < NTOK) {
                    int d0 = ii / WIN - j / WIN + (WIN - 1);
                    int d1 = ii % WIN - j % WIN + (WIN - 1);
                    v += rpb[(d0 * (2 * WIN - 1) + d1) * HEADS + head];
                    if (swin_label(wy, wx, ii) != swin_label(wy, wx, j)) v += -100.f;
                }
                Sw[ii * 65 + j] = v;
            }
        }
    }
    __syncthreads();

    // row softmax (fp32), write P as f16
    #pragma unroll
    for (int rr = 0; rr < 2; ++rr) {
        int i = lane * 2 + rr;
        float mx = -3.4e38f;
        for (int j = 0; j < 64; ++j) mx = fmaxf(mx, Sw[i * 65 + j]);
        float s = 0.f;
        for (int j = 0; j < 64; ++j) {
            float e = __expf(Sw[i * 65 + j] - mx);
            s += e; Sw[i * 65 + j] = e;
        }
        float inv = 1.f / s;
        for (int j = 0; j < 64; ++j)
            Pw[i * 72 + j] = (_Float16)(Sw[i * 65 + j] * inv);
    }
    __syncthreads();

    // O = P @ Vt   (K = 64 -> two WMMA steps of 32)
    v16h vf[2][2];
    #pragma unroll
    for (int nt = 0; nt < 2; ++nt)
        #pragma unroll
        for (int kt = 0; kt < 2; ++kt) {
            const _Float16* p = &Vt[(nt * 16 + m16) * 72 + kt * 32 + kh * 8];
            vf[nt][kt] = cat8(*(const v8h*)p, *(const v8h*)(p + 16));
        }
    #pragma unroll
    for (int mt = 0; mt < 4; ++mt) {
        v16h pf[2];
        #pragma unroll
        for (int kt = 0; kt < 2; ++kt) {
            const _Float16* p = &Pw[(mt * 16 + m16) * 72 + kt * 32 + kh * 8];
            pf[kt] = cat8(*(const v8h*)p, *(const v8h*)(p + 16));
        }
        #pragma unroll
        for (int nt = 0; nt < 2; ++nt) {
            v8f c = z;
            c = wmma16(pf[0], vf[nt][0], c);
            c = wmma16(pf[1], vf[nt][1], c);
            #pragma unroll
            for (int r = 0; r < 8; ++r) {
                int ii = mt * 16 + r + 8 * kh;
                if (ii < NTOK)
                    out[(qbase + ii) * CC + head * HD + nt * 16 + m16] = (_Float16)c[r];
            }
        }
    }
}

// ------------------------------------------------------------------
// LayerNorm (+optional shift/window gather) with f16 cast. 1 row/block.
// ------------------------------------------------------------------
template<bool WINDOWED>
__global__ __launch_bounds__(128) void ln_cast_kernel(
    const float* __restrict__ in, const float* __restrict__ g,
    const float* __restrict__ bta, _Float16* __restrict__ out)
{
    __shared__ float red[128];
    const int tid = threadIdx.x;
    const int row = blockIdx.x;
    const int src = WINDOWED ? win_row_to_orig(row) : row;
    const float* ip = in + (size_t)src * CC;
    float v0 = ip[tid], v1 = ip[tid + 128], v2 = ip[tid + 256];
    red[tid] = v0 + v1 + v2;
    __syncthreads();
    for (int s = 64; s > 0; s >>= 1) { if (tid < s) red[tid] += red[tid + s]; __syncthreads(); }
    float mean = red[0] * (1.f / (float)CC);
    __syncthreads();
    float d0 = v0 - mean, d1 = v1 - mean, d2 = v2 - mean;
    red[tid] = d0 * d0 + d1 * d1 + d2 * d2;
    __syncthreads();
    for (int s = 64; s > 0; s >>= 1) { if (tid < s) red[tid] += red[tid + s]; __syncthreads(); }
    float rstd = rsqrtf(red[0] * (1.f / (float)CC) + 1e-5f);
    _Float16* op = out + (size_t)row * CC;
    op[tid]       = (_Float16)(d0 * rstd * g[tid]       + bta[tid]);
    op[tid + 128] = (_Float16)(d1 * rstd * g[tid + 128] + bta[tid + 128]);
    op[tid + 256] = (_Float16)(d2 * rstd * g[tid + 256] + bta[tid + 256]);
}

// ------------------------------------------------------------------
// Weight transpose + fp32 -> f16 cast: in[R,C] -> out[C,R]
// ------------------------------------------------------------------
__global__ __launch_bounds__(256) void transpose_cast_kernel(
    const float* __restrict__ in, _Float16* __restrict__ out, int R, int C)
{
    int idx = blockIdx.x * 256 + threadIdx.x;
    if (idx < R * C) {
        int r = idx / C, c = idx - r * C;
        out[(size_t)c * R + r] = (_Float16)in[idx];
    }
}

extern "C" void kernel_launch(void* const* d_in, const int* in_sizes, int n_in,
                              void* d_out, int out_size, void* d_ws, size_t ws_size,
                              hipStream_t stream)
{
    (void)in_sizes; (void)n_in; (void)out_size; (void)ws_size;
    const float* x      = (const float*)d_in[0];
    const float* n1g    = (const float*)d_in[1];
    const float* n1b    = (const float*)d_in[2];
    const float* qkv_w  = (const float*)d_in[3];
    const float* qkv_b  = (const float*)d_in[4];
    const float* proj_w = (const float*)d_in[5];
    const float* proj_b = (const float*)d_in[6];
    const float* rpb    = (const float*)d_in[7];
    const float* n2g    = (const float*)d_in[8];
    const float* n2b    = (const float*)d_in[9];
    const float* w1     = (const float*)d_in[10];
    const float* b1     = (const float*)d_in[11];
    const float* w2     = (const float*)d_in[12];
    const float* b2     = (const float*)d_in[13];
    float* out = (float*)d_out;

    char* ws = (char*)d_ws;
    size_t off = 0;
    auto carve = [&](size_t bytes) -> char* {
        char* p = ws + off;
        off = (off + bytes + 255) & ~(size_t)255;
        return p;
    };
    _Float16* wT_qkv = (_Float16*)carve((size_t)1152 * 384 * 2);
    _Float16* wT_prj = (_Float16*)carve((size_t)384  * 384 * 2);
    _Float16* wT_m1  = (_Float16*)carve((size_t)1536 * 384 * 2);
    _Float16* wT_m2  = (_Float16*)carve((size_t)384  * 1536 * 2);
    _Float16* fbuf   = (_Float16*)carve((size_t)MTOT * CC * 2);    // win -> attn_out -> h2
    _Float16* qkvb   = (_Float16*)carve((size_t)MTOT * 1152 * 2);
    float*    x2     = (float*)   carve((size_t)MTOT * CC * 4);
    _Float16* gbuf   = (_Float16*)carve((size_t)MTOT * MLPD * 2);

    // 0) weight transposes + f16 cast
    transpose_cast_kernel<<<(384*1152 + 255)/256, 256, 0, stream>>>(qkv_w,  wT_qkv, 384, 1152);
    transpose_cast_kernel<<<(384*384  + 255)/256, 256, 0, stream>>>(proj_w, wT_prj, 384, 384);
    transpose_cast_kernel<<<(384*1536 + 255)/256, 256, 0, stream>>>(w1,     wT_m1,  384, 1536);
    transpose_cast_kernel<<<(1536*384 + 255)/256, 256, 0, stream>>>(w2,     wT_m2,  1536, 384);

    // 1) LN1 + shift + window partition (window-ordered f16)
    ln_cast_kernel<true><<<MTOT, 128, 0, stream>>>(x, n1g, n1b, fbuf);
    // 2) QKV GEMM (q scaled), f16 out
    gemm_wmma_kernel<EPI_QKV><<<dim3(1152/128, MTOT/128), 256, 0, stream>>>(
        fbuf, wT_qkv, qkv_b, nullptr, qkvb, MTOT, 1152, 384);
    // 3) windowed attention (writes attn_out into fbuf, window-ordered)
    attn_kernel<<<dim3(BB * NWIN, HEADS / 2), 64, 0, stream>>>(qkvb, rpb, fbuf);
    // 4) proj GEMM + window-reverse + unshift + residual(x) -> x2 fp32
    gemm_wmma_kernel<EPI_PROJ><<<dim3(384/128, MTOT/128), 256, 0, stream>>>(
        fbuf, wT_prj, proj_b, x, x2, MTOT, 384, 384);
    // 5) LN2 -> f16 (reuse fbuf)
    ln_cast_kernel<false><<<MTOT, 128, 0, stream>>>(x2, n2g, n2b, fbuf);
    // 6) MLP1 GEMM + exact GELU -> f16
    gemm_wmma_kernel<EPI_MLP1><<<dim3(1536/128, MTOT/128), 256, 0, stream>>>(
        fbuf, wT_m1, b1, nullptr, gbuf, MTOT, 1536, 384);
    // 7) MLP2 GEMM + bias + residual(x2) -> d_out fp32
    gemm_wmma_kernel<EPI_MLP2><<<dim3(384/128, MTOT/128), 256, 0, stream>>>(
        gbuf, wT_m2, b2, x2, out, MTOT, 384, 1536);
}
